// NonLocalAttention_34548716929097
// MI455X (gfx1250) — matrix operand order repaired and checked by reading, not compiled, not run on hardware
//
#include <hip/hip_runtime.h>

typedef __attribute__((ext_vector_type(16))) _Float16 v16h;
typedef __attribute__((ext_vector_type(8)))  _Float16 v8h;
typedef __attribute__((ext_vector_type(8)))  float    v8f;

#define DIM   128
#define HEADS 4
#define HD    32
#define WSZ   8
#define TOPK  16
#define TT    2
#define HH    96
#define WW    96
#define NPIX  (HH * WW)          // 9216
#define NROWS (TT * NPIX)        // 18432
#define MBLK  64                 // rows per GEMM block (4 x 16-row M-tiles)
#define SCALE 0.17677669529663687f   // 32^-0.5

// ---------------------------------------------------------------------------
// f32 -> f16 weight conversion
// ---------------------------------------------------------------------------
__global__ void cvt_f32_to_f16(const float* __restrict__ src,
                               _Float16* __restrict__ dst, int n) {
    int i = blockIdx.x * blockDim.x + threadIdx.x;
    if (i < n) dst[i] = (_Float16)src[i];
}

// ---------------------------------------------------------------------------
// Depthwise 3x3 'SAME' (zero pad) for q/k/v simultaneously.
// Emits GEMM A-matrices [p][c] in f16, p = (t,h,w) row-major.
// ---------------------------------------------------------------------------
__global__ void depthwise3x3(const float* __restrict__ vid,
                             const float* __restrict__ wq,
                             const float* __restrict__ wk,
                             const float* __restrict__ wv,
                             _Float16* __restrict__ aq,
                             _Float16* __restrict__ ak,
                             _Float16* __restrict__ av) {
    int i = blockIdx.x * blockDim.x + threadIdx.x;   // over NROWS*DIM
    if (i >= NROWS * DIM) return;
    int c  = i & (DIM - 1);
    int p  = i >> 7;
    int t  = p / NPIX;
    int hw = p - t * NPIX;
    int h  = hw / WW;
    int w  = hw - h * WW;
    const float* x = vid + (size_t)(t * DIM + c) * NPIX;
    float sq = 0.f, sk = 0.f, sv = 0.f;
    #pragma unroll
    for (int dh = 0; dh < 3; ++dh) {
        int hh = h + dh - 1;
        if (hh < 0 || hh >= HH) continue;
        #pragma unroll
        for (int dw = 0; dw < 3; ++dw) {
            int ww = w + dw - 1;
            if (ww < 0 || ww >= WW) continue;
            float xv = x[hh * WW + ww];
            sq += xv * wq[c * 9 + dh * 3 + dw];
            sk += xv * wk[c * 9 + dh * 3 + dw];
            sv += xv * wv[c * 9 + dh * 3 + dw];
        }
    }
    aq[i] = (_Float16)sq;
    ak[i] = (_Float16)sk;
    av[i] = (_Float16)sv;
}

// ---------------------------------------------------------------------------
// Cooperative LDS staging for one MBLK x 128 GEMM block.
// B is repacked fragment-major: sBf[(kt*8+nt)*32 + lane] is the v16h operand
// for that lane -> single 32B LDS read (2x ds_load_b128).
// Fragment K order is contiguous, K = 16*(l/16)+e, same for A and B (wave32).
// ---------------------------------------------------------------------------
__device__ __forceinline__ void stage_tiles(
        const _Float16* __restrict__ A, const _Float16* __restrict__ W,
        int blkRow0, int tid, v16h* sBf, v8h* sAv) {
    // ---- pack W[o*128+c] into fragment-major B layout -------------------
    // B[k=c][n=o]; fragment element e of lane l (n=l%16, khalf=16*(l/16)):
    //   K = kt*32 + khalf + e.  8 consecutive c -> 8 consecutive e.
    _Float16* sBh = (_Float16*)sBf;
    for (int i = tid; i < (DIM * DIM) / 8; i += 128) {
        int o    = i >> 4;            // 16 8-wide chunks per output row
        int c0   = (i & 15) << 3;
        v8h wchunk = *(const v8h*)(W + o * DIM + c0);
        int kt    = c0 >> 5;
        int rem   = c0 & 31;
        int lhalf = rem >> 4;         // 0 or 1
        int e0    = rem & 15;         // 0 or 8
        int ln    = lhalf * 16 + (o & 15);
        int nt    = o >> 4;
        *(v8h*)(sBh + ((((kt * 8 + nt) * 32 + ln) << 4) + e0)) = wchunk;
    }
    // ---- A tiles: straight vector copy, MBLK rows x 128 K ---------------
    for (int i = tid; i < (MBLK * DIM) / 8; i += 128)
        sAv[i] = *(const v8h*)(A + (size_t)blkRow0 * DIM + i * 8);
    __syncthreads();
}

// One 16x32 WMMA pair for M-tile mt; wave owns N-tiles 2*wave, 2*wave+1.
__device__ __forceinline__ void compute_tile(
        const v16h* sBf, const v8h* sAv, int mt, int lane, int wave,
        v8f& acc0, v8f& acc1) {
    int nt0  = wave * 2, nt1 = nt0 + 1;
    int mrow = (mt << 4) + (lane & 15);
    acc0 = {};
    acc1 = {};
    #pragma unroll
    for (int kt = 0; kt < 4; ++kt) {
        // A fragment: K = 16*(l/16)+e
        int ka = kt * 32 + ((lane >> 4) << 4);
        int abase = (mrow * DIM + ka) >> 3;     // v8h index
        v8h alo = sAv[abase];
        v8h ahi = sAv[abase + 1];
        v16h af = __builtin_shufflevector(alo, ahi,
                    0, 1, 2, 3, 4, 5, 6, 7, 8, 9, 10, 11, 12, 13, 14, 15);
        // B fragments: single 32B read each
        v16h bf0 = sBf[(kt * 8 + nt0) * 32 + lane];
        v16h bf1 = sBf[(kt * 8 + nt1) * 32 + lane];
        acc0 = __builtin_amdgcn_wmma_f32_16x16x32_f16(false, af, false, bf0,
                                                      (short)0, acc0, false, false);
        acc1 = __builtin_amdgcn_wmma_f32_16x16x32_f16(false, af, false, bf1,
                                                      (short)0, acc1, false, false);
    }
}

// ---------------------------------------------------------------------------
// GEMM + bias (+ postscale), plain [row][col] f32 store (q/k/v pointwise)
// ---------------------------------------------------------------------------
__global__ void gemm_qkv(const _Float16* __restrict__ A,
                         const _Float16* __restrict__ W,
                         const float* __restrict__ bias,
                         float* __restrict__ C, float postscale) {
    __shared__ v16h sBf[4 * 8 * 32];        // 32 KB, fragment-major B
    __shared__ v8h  sAv[MBLK * DIM / 8];    // 16 KB A tiles
    int tid = threadIdx.x, lane = tid & 31, wave = tid >> 5;
    int blkRow0 = blockIdx.x * MBLK;
    stage_tiles(A, W, blkRow0, tid, sBf, sAv);

    // D layout: m = r + 8*(lane/16), n = lane%16
    int n  = lane & 15;
    int mh = (lane >> 4) * 8;
    int c0 = wave * 32 + n;
    int c1 = c0 + 16;
    float b0 = bias[c0], b1 = bias[c1];
    #pragma unroll
    for (int mt = 0; mt < MBLK / 16; ++mt) {
        v8f acc0, acc1;
        compute_tile(sBf, sAv, mt, lane, wave, acc0, acc1);
        int row0 = blkRow0 + mt * 16;
        #pragma unroll
        for (int r = 0; r < 8; ++r) {
            int m = mh + r;
            C[(size_t)(row0 + m) * DIM + c0] = (acc0[r] + b0) * postscale;
            C[(size_t)(row0 + m) * DIM + c1] = (acc1[r] + b1) * postscale;
        }
    }
}

// ---------------------------------------------------------------------------
// Final projection GEMM + bias, permuted store into [B,T,C,H,W] output
// ---------------------------------------------------------------------------
__global__ void gemm_proj(const _Float16* __restrict__ A,
                          const _Float16* __restrict__ W,
                          const float* __restrict__ bias,
                          float* __restrict__ out) {
    __shared__ v16h sBf[4 * 8 * 32];
    __shared__ v8h  sAv[MBLK * DIM / 8];
    int tid = threadIdx.x, lane = tid & 31, wave = tid >> 5;
    int blkRow0 = blockIdx.x * MBLK;
    stage_tiles(A, W, blkRow0, tid, sBf, sAv);

    int n  = lane & 15;
    int mh = (lane >> 4) * 8;
    int c0 = wave * 32 + n;
    int c1 = c0 + 16;
    float b0 = bias[c0], b1 = bias[c1];
    #pragma unroll
    for (int mt = 0; mt < MBLK / 16; ++mt) {
        v8f acc0, acc1;
        compute_tile(sBf, sAv, mt, lane, wave, acc0, acc1);
        int row0 = blkRow0 + mt * 16;
        #pragma unroll
        for (int r = 0; r < 8; ++r) {
            int p  = row0 + mh + r;
            int t  = p / NPIX;
            int hw = p - t * NPIX;
            out[(size_t)(t * DIM + c0) * NPIX + hw] = acc0[r] + b0;
            out[(size_t)(t * DIM + c1) * NPIX + hw] = acc1[r] + b1;
        }
    }
}

// ---------------------------------------------------------------------------
// Non-local attention: one thread per (query, head).
// 64 candidate dots (hd=32), stable top-16, softmax, weighted V sum -> f16.
// All per-thread arrays statically indexed => register resident (wave32).
// q/k/v vectors are 128B-aligned -> explicit float4 traffic.
// ---------------------------------------------------------------------------
__device__ __forceinline__ int refl(int i, int n) {
    i = (i < 0) ? -i : i;
    return (i >= n) ? 2 * (n - 1) - i : i;
}

__global__ void nl_attention(const float* __restrict__ qb,
                             const float* __restrict__ kb,
                             const float* __restrict__ vb,
                             _Float16* __restrict__ outb) {
    int idx = blockIdx.x * blockDim.x + threadIdx.x;
    if (idx >= NROWS * HEADS) return;
    int head = idx & (HEADS - 1);
    int p    = idx >> 2;
    int t    = p / NPIX;
    int hw   = p - t * NPIX;
    int h    = hw / WW;
    int w    = hw - h * WW;

    const float4* q4 = (const float4*)(qb + (size_t)p * DIM + head * HD);
    float4 qr[HD / 4];
    #pragma unroll
    for (int j = 0; j < HD / 4; ++j) qr[j] = q4[j];

    float topv[TOPK];
    int   topi[TOPK];
    #pragma unroll
    for (int j = 0; j < TOPK; ++j) { topv[j] = -3.0e38f; topi[j] = 0; }

    int base = t * NPIX;
    for (int dy = 0; dy < WSZ; ++dy) {
        int nh = refl(h + dy - WSZ / 2, HH);
        for (int dx = 0; dx < WSZ; ++dx) {
            int nw = refl(w + dx - WSZ / 2, WW);
            int np = base + nh * WW + nw;
            const float4* k4 = (const float4*)(kb + (size_t)np * DIM + head * HD);
            float dot = 0.f;
            #pragma unroll
            for (int j = 0; j < HD / 4; ++j) {
                float4 kv = k4[j];
                dot += qr[j].x * kv.x + qr[j].y * kv.y +
                       qr[j].z * kv.z + qr[j].w * kv.w;
            }
            if (dot > topv[TOPK - 1]) {
                // register-resident bubble insert (strict > keeps stable ties)
                float pv = dot; int pi = np;
                #pragma unroll
                for (int j = 0; j < TOPK; ++j) {
                    bool ins = pv > topv[j];
                    float tv = topv[j]; int ti = topi[j];
                    topv[j] = ins ? pv : tv;
                    topi[j] = ins ? pi : ti;
                    pv = ins ? tv : pv;
                    pi = ins ? ti : pi;
                }
            }
        }
    }

    // softmax over top-k (sorted descending -> max is topv[0]) + weighted V sum
    float mx = topv[0];
    float s  = 0.f;
    float4 acc[HD / 4];
    #pragma unroll
    for (int j = 0; j < HD / 4; ++j) acc[j] = make_float4(0.f, 0.f, 0.f, 0.f);
    #pragma unroll
    for (int j = 0; j < TOPK; ++j) {
        float a = __expf(topv[j] - mx);
        s += a;
        const float4* v4 = (const float4*)(vb + (size_t)topi[j] * DIM + head * HD);
        #pragma unroll
        for (int d = 0; d < HD / 4; ++d) {
            float4 vv = v4[d];
            acc[d].x += a * vv.x; acc[d].y += a * vv.y;
            acc[d].z += a * vv.z; acc[d].w += a * vv.w;
        }
    }
    float inv = 1.f / s;
    // store as four v8h (32 halves, 16B-aligned destination)
    _Float16* o = outb + (size_t)p * DIM + head * HD;
    #pragma unroll
    for (int j = 0; j < HD / 8; ++j) {
        v8h ov;
        ov[0] = (_Float16)(acc[2 * j].x * inv);
        ov[1] = (_Float16)(acc[2 * j].y * inv);
        ov[2] = (_Float16)(acc[2 * j].z * inv);
        ov[3] = (_Float16)(acc[2 * j].w * inv);
        ov[4] = (_Float16)(acc[2 * j + 1].x * inv);
        ov[5] = (_Float16)(acc[2 * j + 1].y * inv);
        ov[6] = (_Float16)(acc[2 * j + 1].z * inv);
        ov[7] = (_Float16)(acc[2 * j + 1].w * inv);
        *(v8h*)(o + j * 8) = ov;
    }
}

// ---------------------------------------------------------------------------
// Launch
// ---------------------------------------------------------------------------
extern "C" void kernel_launch(void* const* d_in, const int* in_sizes, int n_in,
                              void* d_out, int out_size, void* d_ws, size_t ws_size,
                              hipStream_t stream) {
    const float* vid    = (const float*)d_in[0];
    const float* wq_dw  = (const float*)d_in[1];
    const float* wq_pw  = (const float*)d_in[2];
    const float* bq     = (const float*)d_in[3];
    const float* wk_dw  = (const float*)d_in[4];
    const float* wk_pw  = (const float*)d_in[5];
    const float* bk     = (const float*)d_in[6];
    const float* wv_dw  = (const float*)d_in[7];
    const float* wv_pw  = (const float*)d_in[8];
    const float* bv     = (const float*)d_in[9];
    const float* proj_w = (const float*)d_in[10];
    const float* proj_b = (const float*)d_in[11];
    float* out = (float*)d_out;

    // workspace carve-out
    char* ws = (char*)d_ws;
    size_t off = 0;
    auto carve = [&](size_t bytes) -> void* {
        void* p = ws + off;
        off = (off + bytes + 255) & ~(size_t)255;
        return p;
    };
    _Float16* wq16 = (_Float16*)carve(DIM * DIM * sizeof(_Float16));
    _Float16* wk16 = (_Float16*)carve(DIM * DIM * sizeof(_Float16));
    _Float16* wv16 = (_Float16*)carve(DIM * DIM * sizeof(_Float16));
    _Float16* wp16 = (_Float16*)carve(DIM * DIM * sizeof(_Float16));
    _Float16* aq   = (_Float16*)carve((size_t)NROWS * DIM * sizeof(_Float16));
    _Float16* ak   = (_Float16*)carve((size_t)NROWS * DIM * sizeof(_Float16));
    _Float16* av   = (_Float16*)carve((size_t)NROWS * DIM * sizeof(_Float16));
    float*    qf   = (float*)   carve((size_t)NROWS * DIM * sizeof(float));
    float*    kf   = (float*)   carve((size_t)NROWS * DIM * sizeof(float));
    float*    vf   = (float*)   carve((size_t)NROWS * DIM * sizeof(float));
    _Float16* ao16 = (_Float16*)carve((size_t)NROWS * DIM * sizeof(_Float16));

    // 1) weight conversion to f16
    int wn = DIM * DIM;
    cvt_f32_to_f16<<<(wn + 255) / 256, 256, 0, stream>>>(wq_pw, wq16, wn);
    cvt_f32_to_f16<<<(wn + 255) / 256, 256, 0, stream>>>(wk_pw, wk16, wn);
    cvt_f32_to_f16<<<(wn + 255) / 256, 256, 0, stream>>>(wv_pw, wv16, wn);
    cvt_f32_to_f16<<<(wn + 255) / 256, 256, 0, stream>>>(proj_w, wp16, wn);

    // 2) depthwise 3x3 for q/k/v -> f16 GEMM A matrices
    int dn = NROWS * DIM;
    depthwise3x3<<<(dn + 255) / 256, 256, 0, stream>>>(vid, wq_dw, wk_dw, wv_dw,
                                                       aq, ak, av);

    // 3) pointwise 1x1 as WMMA GEMMs (+bias, q pre-scaled), 64 rows/block
    gemm_qkv<<<NROWS / MBLK, 128, 0, stream>>>(aq, wq16, bq, qf, SCALE);
    gemm_qkv<<<NROWS / MBLK, 128, 0, stream>>>(ak, wk16, bk, kf, 1.0f);
    gemm_qkv<<<NROWS / MBLK, 128, 0, stream>>>(av, wv16, bv, vf, 1.0f);

    // 4) non-local attention (top-16 of 8x8 window, softmax, weighted V)
    int an = NROWS * HEADS;
    nl_attention<<<(an + 255) / 256, 256, 0, stream>>>(qf, kf, vf, ao16);

    // 5) output projection WMMA GEMM + bias, permuted store to [B,T,C,H,W]
    gemm_proj<<<NROWS / MBLK, 128, 0, stream>>>(ao16, wp16, proj_b, out);
}
